// GRU_57346403336196
// MI455X (gfx1250) — matrix operand order repaired and checked by reading, not compiled, not run on hardware
//
#include <hip/hip_runtime.h>

// ---------------- CDNA5 WMMA types ----------------
typedef __attribute__((ext_vector_type(16))) __bf16 v16bf;
typedef __attribute__((ext_vector_type(8)))  __bf16 v8bf;
typedef __attribute__((ext_vector_type(8)))  float  v8f;

// ---------------- problem constants ----------------
constexpr int B_N = 512;        // batch
constexpr int SEQ = 512;        // input sequence length
constexpr int IN  = 128;        // input / output feature dim
constexpr int HID = 1024;       // hidden
constexpr int PRE = 96;         // decoded steps

// round-to-nearest-even f32 -> bf16
static __device__ __forceinline__ __bf16 f2bf(float f) {
  unsigned u = __builtin_bit_cast(unsigned, f);
  unsigned r = (u + 0x7FFFu + ((u >> 16) & 1u)) >> 16;
  unsigned short s = (unsigned short)r;
  return __builtin_bit_cast(__bf16, s);
}

// A fragment: two adjacent aligned 16B chunks -> concat, no elt moves.
static __device__ __forceinline__ v16bf ld_a(const __bf16* p) {
  v8bf lo = *reinterpret_cast<const v8bf*>(p);       // K = k0 + half*16 ..
  v8bf hi = *reinterpret_cast<const v8bf*>(p + 8);   // K = k0+half*16+8 ..
  return __builtin_shufflevector(lo, hi, 0, 1, 2, 3, 4, 5, 6, 7,
                                 8, 9, 10, 11, 12, 13, 14, 15);
}

// B fragment: 32 contiguous, 32B-aligned bytes -> single vector load.
static __device__ __forceinline__ v16bf ld_b(const __bf16* p) {
  return *reinterpret_cast<const v16bf*>(p);
}

static __device__ __forceinline__ v8f wmma_bf16(v16bf a, v16bf b, v8f c) {
  return __builtin_amdgcn_wmma_f32_16x16x32_bf16(false, a, false, b,
                                                 (short)0, c, false, false);
}

// Fence between the load phase and the compute phase of a K-chunk:
// nothing may cross, so all fragment loads issue before the first WMMA
// and the waits become progressive instead of loadcnt==0 drains.
static __device__ __forceinline__ void sched_fence() {
  __builtin_amdgcn_sched_barrier(0);
}

// ---------------------------------------------------------------
// Fully fused GRU step:  h_out = GRUCell(x_t, h_in)
// Grid: 8 m-blocks x 16 j-blocks = 128 blocks, 256 threads (8 waves).
// Wave tile: 16 batch rows x 32 hidden cols; accs {R,Z,HN,IN} x 2.
// ---------------------------------------------------------------
template <bool XF32>
__global__ void __launch_bounds__(256)
gru_step(const __bf16* __restrict__ hBin, const float* __restrict__ hFin,
         const void*   __restrict__ xin,  size_t xStride,
         const __bf16* __restrict__ whh,  const __bf16* __restrict__ wih,
         const float*  __restrict__ b_ih, const float* __restrict__ b_hh,
         float* __restrict__ hFout, __bf16* __restrict__ hBout)
{
  const int tid  = threadIdx.x;
  const int wave = tid >> 5;
  const int lane = tid & 31;
  const int half = lane >> 4;
  const int lm   = lane & 15;

  const int m0 = (blockIdx.x >> 4) * 64 + (wave & 3) * 16;   // 16 batch rows
  const int jb = (blockIdx.x & 15) * 64 + (wave >> 2) * 32;  // 32 hidden cols
  const int c0 = jb + lm;                                    // t=0 column
  const int c1 = jb + 16 + lm;                               // t=1 column

  v8f vz = {};
  v8f accR[2], accZ[2], accHN[2], accIN[2];
#pragma unroll
  for (int t = 0; t < 2; ++t) { accR[t] = vz; accZ[t] = vz; accHN[t] = vz; accIN[t] = vz; }

  // ---- hidden-state GEMM: K = HID ----
  {
    const __bf16* arow = hBin + (size_t)(m0 + lm) * HID + half * 16;
    const __bf16* pR0  = whh + (size_t)c0 * HID + half * 16;
    const __bf16* pZ0  = whh + (size_t)(HID + c0) * HID + half * 16;
    const __bf16* pN0  = whh + (size_t)(2 * HID + c0) * HID + half * 16;
    const __bf16* pR1  = whh + (size_t)c1 * HID + half * 16;
    const __bf16* pZ1  = whh + (size_t)(HID + c1) * HID + half * 16;
    const __bf16* pN1  = whh + (size_t)(2 * HID + c1) * HID + half * 16;

    for (int k0 = 0; k0 < HID; k0 += 32) {
      // ---- load phase: 1 A frag + 6 B frags, distinct registers ----
      v16bf af = ld_a(arow + k0);
      v16bf b0 = ld_b(pR0 + k0);
      v16bf b1 = ld_b(pZ0 + k0);
      v16bf b2 = ld_b(pN0 + k0);
      v16bf b3 = ld_b(pR1 + k0);
      v16bf b4 = ld_b(pZ1 + k0);
      v16bf b5 = ld_b(pN1 + k0);
      sched_fence();   // all loads in flight before first WMMA
      // ---- compute phase: progressive loadcnt waits, loads overlap math ----
      accR[0]  = wmma_bf16(af, b0, accR[0]);
      accZ[0]  = wmma_bf16(af, b1, accZ[0]);
      accHN[0] = wmma_bf16(af, b2, accHN[0]);
      accR[1]  = wmma_bf16(af, b3, accR[1]);
      accZ[1]  = wmma_bf16(af, b4, accZ[1]);
      accHN[1] = wmma_bf16(af, b5, accHN[1]);
    }
  }

  // ---- input GEMM: K = IN (encoder: f32 inline-converted; decoder: bf16) ----
  {
    const __bf16* pR0 = wih + (size_t)c0 * IN + half * 16;
    const __bf16* pZ0 = wih + (size_t)(HID + c0) * IN + half * 16;
    const __bf16* pN0 = wih + (size_t)(2 * HID + c0) * IN + half * 16;
    const __bf16* pR1 = wih + (size_t)c1 * IN + half * 16;
    const __bf16* pZ1 = wih + (size_t)(HID + c1) * IN + half * 16;
    const __bf16* pN1 = wih + (size_t)(2 * HID + c1) * IN + half * 16;

#pragma unroll
    for (int k0 = 0; k0 < IN; k0 += 32) {
      v16bf af;
      if (XF32) {
        const float* ar = (const float*)xin + (size_t)(m0 + lm) * xStride + half * 16;
        v8f lo = *reinterpret_cast<const v8f*>(ar + k0);
        v8f hi = *reinterpret_cast<const v8f*>(ar + k0 + 8);
#pragma unroll
        for (int i = 0; i < 8; ++i) { af[i] = f2bf(lo[i]); af[i + 8] = f2bf(hi[i]); }
      } else {
        af = ld_a((const __bf16*)xin + (size_t)(m0 + lm) * xStride + half * 16 + k0);
      }
      v16bf b0 = ld_b(pR0 + k0);
      v16bf b1 = ld_b(pZ0 + k0);
      v16bf b2 = ld_b(pN0 + k0);
      v16bf b3 = ld_b(pR1 + k0);
      v16bf b4 = ld_b(pZ1 + k0);
      v16bf b5 = ld_b(pN1 + k0);
      sched_fence();
      accR[0]  = wmma_bf16(af, b0, accR[0]);   // i_r adds onto h_r
      accZ[0]  = wmma_bf16(af, b1, accZ[0]);   // i_z adds onto h_z
      accIN[0] = wmma_bf16(af, b2, accIN[0]);  // i_n kept separate
      accR[1]  = wmma_bf16(af, b3, accR[1]);
      accZ[1]  = wmma_bf16(af, b4, accZ[1]);
      accIN[1] = wmma_bf16(af, b5, accIN[1]);
    }
  }

  // ---- fused gate epilogue: sigmoid/tanh + h update straight from accs ----
#pragma unroll
  for (int t = 0; t < 2; ++t) {
    const int col = jb + t * 16 + lm;
    const float bir  = b_ih[col],           bhr = b_hh[col];
    const float biz  = b_ih[HID + col],     bhz = b_hh[HID + col];
    const float bin_ = b_ih[2 * HID + col], bhn = b_hh[2 * HID + col];
#pragma unroll
    for (int r = 0; r < 8; ++r) {
      const int row = m0 + half * 8 + r;   // C/D layout: VGPR r -> row half*8+r
      const float rg = 1.f / (1.f + __expf(-(accR[t][r] + bir + bhr)));
      const float zg = 1.f / (1.f + __expf(-(accZ[t][r] + biz + bhz)));
      const float ng = tanhf(accIN[t][r] + bin_ + rg * (accHN[t][r] + bhn));
      const float hold = hFin[(size_t)row * HID + col];
      const float hnew = (1.f - zg) * ng + zg * hold;
      hFout[(size_t)row * HID + col] = hnew;
      hBout[(size_t)row * HID + col] = f2bf(hnew);
    }
  }
}

// ---------------------------------------------------------------
// FC: out = h @ fc_w^T + fc_b  (M=512, N=128, K=1024)
// Writes f32 slice pred[:,p,:] of d_out AND the bf16 decoder input.
// ---------------------------------------------------------------
__global__ void __launch_bounds__(256)
fc_wmma(const __bf16* __restrict__ hB, const __bf16* __restrict__ fcw,
        const float* __restrict__ fcb, float* __restrict__ out,
        __bf16* __restrict__ xcur, int p)
{
  const int tid  = threadIdx.x;
  const int wave = tid >> 5;
  const int lane = tid & 31;
  const int half = lane >> 4;
  const int lm   = lane & 15;

  const int m0 = blockIdx.x * 64 + (wave & 3) * 16;
  const int nb = (wave >> 2) * 64;          // 0 or 64; N = 128 total

  v8f vz = {};
  v8f acc[4];
#pragma unroll
  for (int t = 0; t < 4; ++t) acc[t] = vz;

  const __bf16* arow = hB + (size_t)(m0 + lm) * HID + half * 16;
  const __bf16* pw0 = fcw + (size_t)(nb + lm) * HID + half * 16;
  const __bf16* pw1 = fcw + (size_t)(nb + 16 + lm) * HID + half * 16;
  const __bf16* pw2 = fcw + (size_t)(nb + 32 + lm) * HID + half * 16;
  const __bf16* pw3 = fcw + (size_t)(nb + 48 + lm) * HID + half * 16;

  for (int k0 = 0; k0 < HID; k0 += 32) {
    v16bf af = ld_a(arow + k0);
    v16bf b0 = ld_b(pw0 + k0);
    v16bf b1 = ld_b(pw1 + k0);
    v16bf b2 = ld_b(pw2 + k0);
    v16bf b3 = ld_b(pw3 + k0);
    sched_fence();
    acc[0] = wmma_bf16(af, b0, acc[0]);
    acc[1] = wmma_bf16(af, b1, acc[1]);
    acc[2] = wmma_bf16(af, b2, acc[2]);
    acc[3] = wmma_bf16(af, b3, acc[3]);
  }

#pragma unroll
  for (int t = 0; t < 4; ++t) {
    const int col  = nb + t * 16 + lm;
    const float bias = fcb[col];
#pragma unroll
    for (int r = 0; r < 8; ++r) {
      const int row = m0 + half * 8 + r;
      const float v = acc[t][r] + bias;
      out[((size_t)row * PRE + p) * IN + col] = v;   // pred[row, p, col]
      xcur[(size_t)row * IN + col] = f2bf(v);        // next decoder input
    }
  }
}

// ---------------- small helpers ----------------
__global__ void cvt_bf16(const float* __restrict__ s, __bf16* __restrict__ d, int n)
{
  const int i = blockIdx.x * blockDim.x + threadIdx.x;
  if (i < n) d[i] = f2bf(s[i]);
}

__global__ void init_h(float* __restrict__ hF, __bf16* __restrict__ hB, int n)
{
  const int i = blockIdx.x * blockDim.x + threadIdx.x;
  if (i < n) { hF[i] = 0.f; hB[i] = f2bf(0.f); }
}

// ---------------------------------------------------------------
extern "C" void kernel_launch(void* const* d_in, const int* in_sizes, int n_in,
                              void* d_out, int out_size, void* d_ws, size_t ws_size,
                              hipStream_t stream)
{
  const float* x    = (const float*)d_in[0];  // [B,S,I]
  const float* w_ih = (const float*)d_in[1];  // [3H,I]
  const float* w_hh = (const float*)d_in[2];  // [3H,H]
  const float* b_ih = (const float*)d_in[3];  // [3H]
  const float* b_hh = (const float*)d_in[4];  // [3H]
  const float* fc_w = (const float*)d_in[5];  // [I,H]
  const float* fc_b = (const float*)d_in[6];  // [I]
  float* out = (float*)d_out;                 // [B,PRE,I]
  (void)in_sizes; (void)n_in; (void)out_size; (void)ws_size;

  char* ws = (char*)d_ws;
  size_t off = 0;
  auto carve = [&](size_t bytes) -> char* {
    char* p = ws + off; off += (bytes + 255) & ~size_t(255); return p;
  };
  __bf16* whh_b  = (__bf16*)carve((size_t)3 * HID * HID * 2);
  __bf16* wih_b  = (__bf16*)carve((size_t)3 * HID * IN  * 2);
  __bf16* fcw_b  = (__bf16*)carve((size_t)IN * HID * 2);
  float*  hF[2]  = { (float*)carve((size_t)B_N * HID * 4),
                     (float*)carve((size_t)B_N * HID * 4) };
  __bf16* hBf[2] = { (__bf16*)carve((size_t)B_N * HID * 2),
                     (__bf16*)carve((size_t)B_N * HID * 2) };
  __bf16* xcur   = (__bf16*)carve((size_t)B_N * IN * 2);

  // per-call weight conversion + h0 = 0 (only the step-0 input buffer)
  cvt_bf16<<<(3 * HID * HID + 255) / 256, 256, 0, stream>>>(w_hh, whh_b, 3 * HID * HID);
  cvt_bf16<<<(3 * HID * IN  + 255) / 256, 256, 0, stream>>>(w_ih, wih_b, 3 * HID * IN);
  cvt_bf16<<<(IN * HID + 255) / 256, 256, 0, stream>>>(fc_w, fcw_b, IN * HID);
  init_h  <<<(B_N * HID + 255) / 256, 256, 0, stream>>>(hF[0], hBf[0], B_N * HID);

  const int STEP_BLOCKS = (B_N / 64) * (HID / 64);  // 8 * 16 = 128

  // encoder: one fused kernel per timestep (ping-pong h buffers)
  int cur = 0;
  for (int t = 0; t < SEQ; ++t) {
    gru_step<true><<<STEP_BLOCKS, 256, 0, stream>>>(
        hBf[cur], hF[cur], (const void*)(x + (size_t)t * IN), (size_t)SEQ * IN,
        whh_b, wih_b, b_ih, b_hh, hF[cur ^ 1], hBf[cur ^ 1]);
    cur ^= 1;
  }

  // out0 = h @ fc_w^T + fc_b -> pred[:,0,:] and decoder input (bf16)
  fc_wmma<<<B_N / 64, 256, 0, stream>>>(hBf[cur], fcw_b, fc_b, out, xcur, 0);

  // autoregressive decoder: fused GRU step + FC per step
  for (int p = 1; p < PRE; ++p) {
    gru_step<false><<<STEP_BLOCKS, 256, 0, stream>>>(
        hBf[cur], hF[cur], (const void*)xcur, (size_t)IN,
        whh_b, wih_b, b_ih, b_hh, hF[cur ^ 1], hBf[cur ^ 1]);
    cur ^= 1;
    fc_wmma<<<B_N / 64, 256, 0, stream>>>(hBf[cur], fcw_b, fc_b, out, xcur, p);
  }
}
